// axis_aligned_gmm_layer_87995289960656
// MI455X (gfx1250) — matrix operand order repaired and checked by reading, not compiled, not run on hardware
//
#include <hip/hip_runtime.h>

// Problem constants (match reference).
#define BB    8192
#define DIMN  1024
#define KK    256
#define NCH   64     // chunks of 16 d-elements (= 32 interleaved kd)
#define MBLK  64     // batch rows per block
#define NSUB  4      // 16-row M-subtiles per block

typedef __attribute__((ext_vector_type(8)))  float  v8f;
typedef __bf16 v16bf __attribute__((ext_vector_type(16)));

union Frag {              // 8 VGPRs = 16 bf16 = one WMMA A or B operand
  unsigned u[8];
  uint4    q[2];
  v16bf    v;
};

// ---- bf16 helpers: round-to-nearest-even hi, residual lo ----
__device__ __forceinline__ unsigned short bf16_rn(float f) {
  unsigned u = __float_as_uint(f);
  u += 0x7FFFu + ((u >> 16) & 1u);
  return (unsigned short)(u >> 16);
}
__device__ __forceinline__ float bf16_f32(unsigned short h) {
  return __uint_as_float(((unsigned)h) << 16);
}
__device__ __forceinline__ unsigned pack2(unsigned short lo, unsigned short hi) {
  return (unsigned)lo | ((unsigned)hi << 16);
}

// =====================================================================
// Kernel 1: per-class preprocessing.
//   b0 = 1/(|D|+1e-4)          (coefficient of x^2 channel, kd even)
//   b1 = -2*c/(|D|+1e-4)       (coefficient of x   channel, kd odd)
// packed directly into the WMMA B-matrix lane layout (32x16 bf16 tile):
//   lanes 0-15  = columns, kd 0..15  (2 per dword, ascending)
//   lanes 16-31 = columns, kd 16..31
// Also reduces log_det[k] and c2[k] = sum c^2 * Dinv.
// =====================================================================
__global__ void gmm_prep_k(const float* __restrict__ centers,
                           const float* __restrict__ Dm,
                           unsigned* __restrict__ Bh, unsigned* __restrict__ Bl,
                           float* __restrict__ c2,   float* __restrict__ logdet) {
  const int k = blockIdx.x;      // class
  const int t = threadIdx.x;     // 256 threads, 4 d each
  float lsum = 0.f, csum = 0.f;
  for (int i = 0; i < 4; ++i) {
    const int d   = t + 256 * i;
    const float dv   = fabsf(Dm[k * DIMN + d]) + 1e-4f;
    const float dinv = 1.0f / dv;
    lsum += logf(dv);
    const float c = centers[k * DIMN + d];
    csum += c * c * dinv;
    const float b0 = dinv;
    const float b1 = -2.0f * c * dinv;
    const unsigned short b0h = bf16_rn(b0);
    const unsigned short b1h = bf16_rn(b1);
    const unsigned short b0l = bf16_rn(b0 - bf16_f32(b0h));
    const unsigned short b1l = bf16_rn(b1 - bf16_f32(b1h));
    // B-fragment lane layout
    const int tt = k >> 4, lanelow = k & 15;
    const int cc = d >> 4, dl = d & 15;
    const int lane = ((dl >> 3) << 4) + lanelow;  // K-half in lane[4]
    const int j    = dl & 7;                      // dword index within lane
    const size_t idx = ((size_t)(cc * 16 + tt) * 32 + lane) * 8 + j;
    Bh[idx] = pack2(b0h, b1h);   // low16 = even kd (x^2), high16 = odd kd (x)
    Bl[idx] = pack2(b0l, b1l);
  }
  __shared__ float red[256];
  red[t] = lsum; __syncthreads();
  for (int s = 128; s > 0; s >>= 1) { if (t < s) red[t] += red[t + s]; __syncthreads(); }
  if (t == 0) logdet[k] = red[0];
  __syncthreads();
  red[t] = csum; __syncthreads();
  for (int s = 128; s > 0; s >>= 1) { if (t < s) red[t] += red[t + s]; __syncthreads(); }
  if (t == 0) c2[k] = red[0];
}

// =====================================================================
// Kernel 2: det_scale_safe[k] = exp(-0.5*logdet[k] - max_k(-0.5*logdet))
// =====================================================================
__global__ void gmm_prep_scale(const float* __restrict__ logdet,
                               float* __restrict__ dscale) {
  const int k = threadIdx.x;               // 256 threads
  const float f = -0.5f * logdet[k];
  __shared__ float red[256];
  red[k] = f; __syncthreads();
  for (int s = 128; s > 0; s >>= 1) { if (k < s) red[k] = fmaxf(red[k], red[k + s]); __syncthreads(); }
  dscale[k] = __expf(f - red[0]);
}

// =====================================================================
// Kernel 3: fused GEMM (bf16 hi/lo x3 WMMA) + weighted row-softmax.
// Block: 256 threads = 8 waves. Block tile: M=64 rows x N=256 classes.
// Wave w owns N-tiles {2w, 2w+1} for all 4 M-subtiles (acc = 8 x v8f).
// =====================================================================
__global__ void __launch_bounds__(256, 1)
gmm_main(const float* __restrict__ x,
         const unsigned* __restrict__ Bh, const unsigned* __restrict__ Bl,
         const float* __restrict__ c2,   const float* __restrict__ dscale,
         float* __restrict__ out) {
  __shared__ __align__(16) unsigned Ah_lds[NSUB * 32 * 8];   // 4 KB
  __shared__ __align__(16) unsigned Al_lds[NSUB * 32 * 8];   // 4 KB
  __shared__ float smax[8 * MBLK];                           // 2 KB
  __shared__ float ssum[8 * MBLK];                           // 2 KB

  const int tid     = threadIdx.x;
  const int lane    = tid & 31;
  const int w       = tid >> 5;
  const int lanelow = lane & 15;
  const int half    = lane >> 4;
  const int rowBase = blockIdx.x * MBLK;
  const int t0 = 2 * w, t1 = 2 * w + 1;

  v8f acc[NSUB][2];
  const v8f vzero = {0.f, 0.f, 0.f, 0.f, 0.f, 0.f, 0.f, 0.f};
  for (int ms = 0; ms < NSUB; ++ms) { acc[ms][0] = vzero; acc[ms][1] = vzero; }

  const uint4* __restrict__ Bh4 = (const uint4*)Bh;
  const uint4* __restrict__ Bl4 = (const uint4*)Bl;
  const uint4* __restrict__ Ah4 = (const uint4*)Ah_lds;
  const uint4* __restrict__ Al4 = (const uint4*)Al_lds;

  // Conversion-thread mapping: row = tid/4, 4 consecutive d per thread.
  const int cr  = tid >> 2;         // 0..63
  const int cd0 = (tid & 3) * 4;    // 0,4,8,12

  for (int cc = 0; cc < NCH; ++cc) {
    // ---- cooperative: x tile -> packed bf16 hi/lo A fragments in LDS ----
    {
      const float4 xv4 = *(const float4*)&x[(size_t)(rowBase + cr) * DIMN + cc * 16 + cd0];
      const float xs[4] = {xv4.x, xv4.y, xv4.z, xv4.w};
      const int msub = cr >> 4, rlow = cr & 15;
      #pragma unroll
      for (int i = 0; i < 4; ++i) {
        const int dl = cd0 + i;
        const float xv = xs[i];
        const float x2 = xv * xv;
        const unsigned short x2h = bf16_rn(x2);
        const unsigned short xvh = bf16_rn(xv);
        const unsigned short x2l = bf16_rn(x2 - bf16_f32(x2h));
        const unsigned short xvl = bf16_rn(xv - bf16_f32(xvh));
        // A-fragment lane layout (16x32 bf16, contiguous k order matching B: kd = 16*half + s)
        const int ahalf = dl >> 3;
        const int j     = dl & 7;
        const int idx   = ((msub * 32 + (ahalf << 4) + rlow) << 3) + j;
        Ah_lds[idx] = pack2(x2h, xvh);   // low16 = even kd (x^2), high16 = odd kd (x)
        Al_lds[idx] = pack2(x2l, xvl);
      }
    }

    // ---- per-wave B fragments for its two N-tiles (L2-resident, 2 MB set) ----
    Frag b0h, b0l, b1h, b1l;
    {
      const size_t bi0 = ((size_t)(cc * 16 + t0) * 32 + lane) * 2;
      const size_t bi1 = ((size_t)(cc * 16 + t1) * 32 + lane) * 2;
      b0h.q[0] = Bh4[bi0]; b0h.q[1] = Bh4[bi0 + 1];
      b0l.q[0] = Bl4[bi0]; b0l.q[1] = Bl4[bi0 + 1];
      b1h.q[0] = Bh4[bi1]; b1h.q[1] = Bh4[bi1 + 1];
      b1l.q[0] = Bl4[bi1]; b1l.q[1] = Bl4[bi1 + 1];
    }

    __syncthreads();

    // ---- 3-WMMA hi/lo accumulation per M-subtile per N-tile ----
    #pragma unroll
    for (int ms = 0; ms < NSUB; ++ms) {
      Frag ah, al;
      const int ai = (ms * 32 + lane) * 2;
      ah.q[0] = Ah4[ai]; ah.q[1] = Ah4[ai + 1];
      al.q[0] = Al4[ai]; al.q[1] = Al4[ai + 1];

      acc[ms][0] = __builtin_amdgcn_wmma_f32_16x16x32_bf16(false, ah.v, false, b0h.v, (short)0, acc[ms][0], false, false);
      acc[ms][0] = __builtin_amdgcn_wmma_f32_16x16x32_bf16(false, al.v, false, b0h.v, (short)0, acc[ms][0], false, false);
      acc[ms][0] = __builtin_amdgcn_wmma_f32_16x16x32_bf16(false, ah.v, false, b0l.v, (short)0, acc[ms][0], false, false);

      acc[ms][1] = __builtin_amdgcn_wmma_f32_16x16x32_bf16(false, ah.v, false, b1h.v, (short)0, acc[ms][1], false, false);
      acc[ms][1] = __builtin_amdgcn_wmma_f32_16x16x32_bf16(false, al.v, false, b1h.v, (short)0, acc[ms][1], false, false);
      acc[ms][1] = __builtin_amdgcn_wmma_f32_16x16x32_bf16(false, ah.v, false, b1l.v, (short)0, acc[ms][1], false, false);
    }
    __syncthreads();
  }

  // ---- fused weighted softmax over K=256 per row ----
  // C/D layout: VGPR r -> row (msub*16 + r + 8*half), column = tile*16 + lanelow.
  const float c2v0 = c2[t0 * 16 + lanelow];
  const float c2v1 = c2[t1 * 16 + lanelow];
  const float ds0  = dscale[t0 * 16 + lanelow];
  const float ds1  = dscale[t1 * 16 + lanelow];

  // Phase A: per-row local max over this wave's 32 columns -> smax.
  #pragma unroll
  for (int ms = 0; ms < NSUB; ++ms) {
    #pragma unroll
    for (int r = 0; r < 8; ++r) {
      const float e0 = -0.5f * (acc[ms][0][r] + c2v0);
      const float e1 = -0.5f * (acc[ms][1][r] + c2v1);
      float mx = fmaxf(e0, e1);
      mx = fmaxf(mx, __shfl_xor(mx, 1, 32));
      mx = fmaxf(mx, __shfl_xor(mx, 2, 32));
      mx = fmaxf(mx, __shfl_xor(mx, 4, 32));
      mx = fmaxf(mx, __shfl_xor(mx, 8, 32));
      if (lanelow == 0) smax[w * MBLK + ms * 16 + r + 8 * half] = mx;
    }
  }
  __syncthreads();

  // Phase B: global row max, weighted exp, per-wave partial sums -> ssum.
  #pragma unroll
  for (int ms = 0; ms < NSUB; ++ms) {
    #pragma unroll
    for (int r = 0; r < 8; ++r) {
      const int m = ms * 16 + r + 8 * half;
      float g = smax[m];
      #pragma unroll
      for (int w2 = 1; w2 < 8; ++w2) g = fmaxf(g, smax[w2 * MBLK + m]);
      const float e0 = -0.5f * (acc[ms][0][r] + c2v0);
      const float e1 = -0.5f * (acc[ms][1][r] + c2v1);
      float s = ds0 * __expf(e0 - g) + ds1 * __expf(e1 - g);
      s += __shfl_xor(s, 1, 32);
      s += __shfl_xor(s, 2, 32);
      s += __shfl_xor(s, 4, 32);
      s += __shfl_xor(s, 8, 32);
      if (lanelow == 0) ssum[w * MBLK + m] = s;
    }
  }
  __syncthreads();

  // Phase C: normalize and store.
  #pragma unroll
  for (int ms = 0; ms < NSUB; ++ms) {
    #pragma unroll
    for (int r = 0; r < 8; ++r) {
      const int m = ms * 16 + r + 8 * half;
      float g = smax[m];
      float tot = ssum[m];
      #pragma unroll
      for (int w2 = 1; w2 < 8; ++w2) {
        g   = fmaxf(g, smax[w2 * MBLK + m]);
        tot += ssum[w2 * MBLK + m];
      }
      const float inv = 1.0f / tot;
      const float e0 = -0.5f * (acc[ms][0][r] + c2v0);
      const float e1 = -0.5f * (acc[ms][1][r] + c2v1);
      const size_t row = (size_t)(rowBase + m) * KK;
      out[row + t0 * 16 + lanelow] = ds0 * __expf(e0 - g) * inv;
      out[row + t1 * 16 + lanelow] = ds1 * __expf(e1 - g) * inv;
    }
  }
}

// =====================================================================
// Host launcher. Workspace layout (needs ~2.01 MiB):
//   [0, 1 MiB)      Bh  packed bf16-hi B' fragments
//   [1 MiB, 2 MiB)  Bl  packed bf16-lo B' fragments
//   [2 MiB, ...)    logdet[256], c2[256], dscale[256]
// =====================================================================
extern "C" void kernel_launch(void* const* d_in, const int* in_sizes, int n_in,
                              void* d_out, int out_size, void* d_ws, size_t ws_size,
                              hipStream_t stream) {
  (void)in_sizes; (void)n_in; (void)out_size; (void)ws_size;
  const float* x       = (const float*)d_in[0];
  const float* centers = (const float*)d_in[1];
  const float* Dm      = (const float*)d_in[2];
  float* out = (float*)d_out;

  char* ws = (char*)d_ws;
  unsigned* Bh   = (unsigned*)ws;                   // 1 MiB
  unsigned* Bl   = (unsigned*)(ws + (1u << 20));    // 1 MiB
  float* logdet  = (float*)(ws + (2u << 20));
  float* c2      = logdet + KK;
  float* dscale  = c2 + KK;

  gmm_prep_k<<<KK, 256, 0, stream>>>(centers, Dm, Bh, Bl, c2, logdet);
  gmm_prep_scale<<<1, KK, 0, stream>>>(logdet, dscale);
  gmm_main<<<BB / MBLK, 256, 0, stream>>>(x, Bh, Bl, c2, dscale, out);
}
